// DAWNLayer_7361573945460
// MI455X (gfx1250) — compile-verified, matrix operand order checked by reading, not run on hardware
//
#include <hip/hip_runtime.h>
#include <math.h>

// ---------------- types ----------------
typedef unsigned short u16;
typedef __attribute__((ext_vector_type(16))) __bf16 bf16x16;
typedef __attribute__((ext_vector_type(8)))  float  f32x8;
typedef __attribute__((ext_vector_type(4)))  unsigned int u32x4;

union Frag { bf16x16 v; u32x4 q[2]; };

// ---------------- gfx1250 async-to-LDS support (guarded) ----------------
#ifndef __has_builtin
#define __has_builtin(x) 0
#endif

#if __has_builtin(__builtin_amdgcn_global_load_async_to_lds_b128)
#define HAVE_ASYNC 1
#else
#define HAVE_ASYNC 0
#endif

#if HAVE_ASYNC
#if __has_builtin(__builtin_amdgcn_s_wait_asynccnt)
#define WAIT_ASYNC(n) __builtin_amdgcn_s_wait_asynccnt(n)
#else
#define WAIT_ASYNC(n)                                                  \
  do {                                                                 \
    if ((n) == 0) asm volatile("s_wait_asynccnt 0x0" ::: "memory");    \
    else          asm volatile("s_wait_asynccnt 0x2" ::: "memory");    \
  } while (0)
#endif

typedef int v4i_vs __attribute__((vector_size(16)));
typedef __attribute__((address_space(1))) v4i_vs* async_gp_t;
typedef __attribute__((address_space(3))) v4i_vs* async_lp_t;

__device__ __forceinline__ void async_copy_b128(const void* g, void* l) {
  __builtin_amdgcn_global_load_async_to_lds_b128(
      (async_gp_t)const_cast<void*>(g), (async_lp_t)l, 0, 0);
}
#endif

__device__ __forceinline__ f32x8 zero8() {
  f32x8 z;
#pragma unroll
  for (int i = 0; i < 8; ++i) z[i] = 0.0f;
  return z;
}

__device__ __forceinline__ u16 f2bf(float f) {
  unsigned int u = __float_as_uint(f);
  u += 0x7fffu + ((u >> 16) & 1u);
  return (u16)(u >> 16);
}

__device__ __forceinline__ f32x8 wmma_bf16(const Frag& a, const Frag& b, f32x8 c) {
  return __builtin_amdgcn_wmma_f32_16x16x32_bf16(false, a.v, false, b.v,
                                                 (short)0, c, false, false);
}

// problem constants
#define BB 2
#define SS 1024
#define DD 1024
#define HH 16
#define DHH 64
#define TOPK 8
#define NNEUR 64
#define NBASIS 32
#define RR 64
#define DFF 4096
#define TT (BB * SS)   // 2048 tokens

// ---------------- small prep kernels ----------------
__global__ void recipe_softmax_kernel(const float* __restrict__ nr, float* __restrict__ smr) {
  int r = threadIdx.x;
  if (r >= NNEUR) return;
  float v[NBASIS];
  float mx = -3.0e38f;
  for (int i = 0; i < NBASIS; ++i) { v[i] = nr[r * NBASIS + i]; mx = fmaxf(mx, v[i]); }
  float s = 0.f;
  for (int i = 0; i < NBASIS; ++i) { v[i] = __expf(v[i] - mx); s += v[i]; }
  float inv = 1.0f / s;
  for (int i = 0; i < NBASIS; ++i) smr[r * NBASIS + i] = v[i] * inv;
}

__global__ void neuron_emb_kernel(const float* __restrict__ smr, const float* __restrict__ bemb,
                                  float* __restrict__ ne) {
  int n = blockIdx.x;
  int t = threadIdx.x;
  __shared__ float w[NBASIS];
  if (t < NBASIS) w[t] = smr[n * NBASIS + t];
  __syncthreads();
  for (int d = t; d < DD; d += 256) {
    float a = 0.f;
#pragma unroll
    for (int b = 0; b < NBASIS; ++b) a += w[b] * bemb[b * DD + d];
    ne[n * DD + d] = a;
  }
}

// generic f32 -> bf16 conversion
__global__ void cvt_bf16_kernel(const float* __restrict__ in, u16* __restrict__ out, int n) {
  for (int i = blockIdx.x * blockDim.x + threadIdx.x; i < n; i += gridDim.x * blockDim.x)
    out[i] = f2bf(in[i]);
}

// basis_A [N][D][R] f32 -> A_pk [D][N*R] bf16 and A_pkT [N*R][D] bf16
__global__ void pack_basis_kernel(const float* __restrict__ bA, u16* __restrict__ apk,
                                  u16* __restrict__ apkT) {
  const int total = NBASIS * DD * RR;
  for (int i = blockIdx.x * blockDim.x + threadIdx.x; i < total; i += gridDim.x * blockDim.x) {
    int n = i / (DD * RR);
    int rem = i - n * (DD * RR);
    int d = rem / RR;
    int r = rem - d * RR;
    u16 v = f2bf(bA[i]);
    apk[(size_t)d * (NBASIS * RR) + n * RR + r] = v;
    apkT[(size_t)(n * RR + r) * DD + d] = v;
  }
}

// ---------------- layernorm ----------------
__device__ __forceinline__ float block_sum(float v, float* red) {
  int t = threadIdx.x;
  red[t] = v;
  __syncthreads();
  for (int s = 128; s > 0; s >>= 1) {
    if (t < s) red[t] += red[t + s];
    __syncthreads();
  }
  float r = red[0];
  __syncthreads();
  return r;
}

__global__ __launch_bounds__(256) void ln_kernel(const float* __restrict__ x,
                                                 const float* __restrict__ g,
                                                 const float* __restrict__ be,
                                                 float* __restrict__ xf,
                                                 u16* __restrict__ xb) {
  __shared__ float red[256];
  int m = blockIdx.x, t = threadIdx.x;
  const float* xr = x + (size_t)m * DD;
  float v[4];
  float s = 0.f;
#pragma unroll
  for (int j = 0; j < 4; ++j) { v[j] = xr[t + j * 256]; s += v[j]; }
  float mean = block_sum(s, red) * (1.0f / DD);
  float s2 = 0.f;
#pragma unroll
  for (int j = 0; j < 4; ++j) { float d = v[j] - mean; s2 += d * d; }
  float var = block_sum(s2, red) * (1.0f / DD);
  float rs = rsqrtf(var + 1e-5f);
#pragma unroll
  for (int j = 0; j < 4; ++j) {
    int idx = t + j * 256;
    float y = (v[j] - mean) * rs * g[idx] + be[idx];
    xf[(size_t)m * DD + idx] = y;
    xb[(size_t)m * DD + idx] = f2bf(y);
  }
}

// ---------------- WMMA GEMM: C(MxN) = A(MxK) * B(KxN) + epilogue ----------------
enum { EPI_BIAS_BF16 = 0, EPI_F32 = 1, EPI_BIAS_ADD_F32 = 2, EPI_XFILT_BF16 = 3, EPI_GELU_BF16 = 4 };

#define ASTRIDE 40   // u16 elements per LDS tile row (32 data + pad, keeps 16B alignment)

template <int MODE>
__global__ __launch_bounds__(256) void gemm_bf16_kernel(
    const u16* __restrict__ A, const u16* __restrict__ B,
    const float* __restrict__ bias, const float* __restrict__ add,
    const float* __restrict__ alphaPtr, void* __restrict__ out,
    int M, int N, int K) {
  __shared__ u16 Al[2][128 * ASTRIDE];
  __shared__ u16 Bl[2][128 * ASTRIDE];

  const int n0 = blockIdx.x * 128;
  const int m0 = blockIdx.y * 128;
  const int t = threadIdx.x;
  const int lane = t & 31;
  const int wave = t >> 5;
  const int wm = wave >> 1;          // 0..3 -> 32-row strips
  const int wn = wave & 1;           // 0..1 -> 64-col strips
  const int lr = lane & 15;
  const int hi = lane >> 4;
  const int halfk = hi * 8;

  f32x8 acc[2][4];
#pragma unroll
  for (int i = 0; i < 2; ++i)
#pragma unroll
    for (int j = 0; j < 4; ++j) acc[i][j] = zero8();

  // ---- staging helpers (A: async b128 when available; B: manual transpose) ----
  auto stageA = [&](int buf, int k0) {
    int c = t;
#pragma unroll
    for (int it = 0; it < 2; ++it, c += 256) {
      int row = c >> 2;
      int k8 = (c & 3) << 3;
#if HAVE_ASYNC
      async_copy_b128(&A[(size_t)(m0 + row) * K + k0 + k8],
                      &Al[buf][row * ASTRIDE + k8]);
#else
      *(u32x4*)&Al[buf][row * ASTRIDE + k8] =
          *(const u32x4*)&A[(size_t)(m0 + row) * K + k0 + k8];
#endif
    }
  };
  auto stageB = [&](int buf, int k0) {
#pragma unroll
    for (int it = 0; it < 16; ++it) {
      int i = t + it * 256;
      int kk = i >> 7;
      int n = i & 127;
      Bl[buf][n * ASTRIDE + kk] = B[(size_t)(k0 + kk) * N + n0 + n];
    }
  };

  const int nk = K >> 5;
  stageA(0, 0);
  stageB(0, 0);

  for (int kt = 0; kt < nk; ++kt) {
    const int cur = kt & 1;
    const bool more = (kt + 1) < nk;
    if (more) {
      stageA(1 - cur, (kt + 1) * 32);
      stageB(1 - cur, (kt + 1) * 32);
      __builtin_prefetch(&A[(size_t)(m0 + (t >> 1)) * K + (kt + 2) * 32], 0, 1);
      __builtin_prefetch(&B[(size_t)((kt + 2) * 32 + (t >> 3)) * N + n0 + (t & 7) * 16], 0, 1);
    }
#if HAVE_ASYNC
    if (more) WAIT_ASYNC(2);   // current tile's 2 per-thread async loads done
    else      WAIT_ASYNC(0);
#endif
    __syncthreads();

    Frag a[2], b[4];
#pragma unroll
    for (int i = 0; i < 2; ++i) {
      const u16* p = &Al[cur][(wm * 32 + i * 16 + lr) * ASTRIDE];
      a[i].q[0] = *(const u32x4*)(p + halfk);
      a[i].q[1] = *(const u32x4*)(p + halfk + 16);
    }
#pragma unroll
    for (int j = 0; j < 4; ++j) {
      const u16* p = &Bl[cur][(wn * 64 + j * 16 + lr) * ASTRIDE];
      b[j].q[0] = *(const u32x4*)(p + halfk);
      b[j].q[1] = *(const u32x4*)(p + halfk + 16);
    }
#pragma unroll
    for (int i = 0; i < 2; ++i)
#pragma unroll
      for (int j = 0; j < 4; ++j) acc[i][j] = wmma_bf16(a[i], b[j], acc[i][j]);
    __syncthreads();
  }

  float alphaV = 0.f;
  if constexpr (MODE == EPI_XFILT_BF16) alphaV = *alphaPtr;

#pragma unroll
  for (int i = 0; i < 2; ++i) {
    int mb = m0 + wm * 32 + i * 16 + 8 * hi;
#pragma unroll
    for (int j = 0; j < 4; ++j) {
      int nb = n0 + wn * 64 + j * 16 + lr;
      float bz = 0.f;
      if constexpr (MODE == EPI_BIAS_BF16 || MODE == EPI_BIAS_ADD_F32 || MODE == EPI_GELU_BF16)
        bz = bias[nb];
#pragma unroll
      for (int e = 0; e < 8; ++e) {
        size_t idx = (size_t)(mb + e) * N + nb;
        float v = acc[i][j][e] + bz;
        if constexpr (MODE == EPI_BIAS_BF16) {
          ((u16*)out)[idx] = f2bf(v);
        } else if constexpr (MODE == EPI_F32) {
          ((float*)out)[idx] = v;
        } else if constexpr (MODE == EPI_BIAS_ADD_F32) {
          ((float*)out)[idx] = v + add[idx];
        } else if constexpr (MODE == EPI_XFILT_BF16) {
          ((u16*)out)[idx] = f2bf(add[idx] + alphaV * acc[i][j][e]);
        } else if constexpr (MODE == EPI_GELU_BF16) {
          float gl = 0.5f * v * (1.0f + erff(v * 0.70710678f));
          ((u16*)out)[idx] = f2bf(gl);
        }
      }
    }
  }
}

// ---------------- flash attention with WMMA ----------------
#define KSTR 72   // Kl row stride (u16), 16B-aligned rows
#define VSTR 40   // Vl / Pl row stride

__global__ __launch_bounds__(256) void attn_kernel(const u16* __restrict__ qb,
                                                   const u16* __restrict__ kb,
                                                   const u16* __restrict__ vb,
                                                   u16* __restrict__ ctx_bf) {
  const int b = blockIdx.x >> 4;   // H = 16
  const int h = blockIdx.x & 15;
  const int qt = blockIdx.y;
  const int t = threadIdx.x;
  const int lane = t & 31;
  const int wave = t >> 5;
  const int lr = lane & 15;
  const int hi = lane >> 4;
  const int halfk = hi * 8;

  __shared__ u16 Kl[32 * KSTR];
  __shared__ u16 Vl[64 * VSTR];
  __shared__ u16 Pl[8 * 16 * VSTR];

  const int qrow = qt * 128 + wave * 16;

  // Q A-fragments (16 rows x 64 dh) straight from global
  Frag aq[2];
  {
    const u16* qp = qb + ((size_t)b * SS + qrow + lr) * DD + h * DHH;
    aq[0].q[0] = *(const u32x4*)(qp + halfk);
    aq[0].q[1] = *(const u32x4*)(qp + halfk + 16);
    aq[1].q[0] = *(const u32x4*)(qp + 32 + halfk);
    aq[1].q[1] = *(const u32x4*)(qp + 32 + halfk + 16);
  }

  f32x8 o[4];
#pragma unroll
  for (int c = 0; c < 4; ++c) o[c] = zero8();
  float m8[8], l8[8];
#pragma unroll
  for (int e = 0; e < 8; ++e) { m8[e] = -3.0e38f; l8[e] = 0.f; }

  const float scale = 0.125f;  // 1/sqrt(64)

  for (int kt0 = 0; kt0 < SS; kt0 += 32) {
#if HAVE_ASYNC
    {
      // K tile: 32 rows x 64 cols = 256 x 16B chunks -> one async b128 per thread
      int key = t >> 3, dh8 = (t & 7) << 3;
      size_t g = ((size_t)b * SS + kt0 + key) * DD + h * DHH + dh8;
      async_copy_b128(&kb[g], &Kl[key * KSTR + dh8]);
      // V tile staged transposed (element-wise, manual)
#pragma unroll
      for (int it = 0; it < 8; ++it) {
        int i = t + it * 256;
        int vkey = i >> 6;
        int dh = i & 63;
        Vl[dh * VSTR + vkey] = vb[((size_t)b * SS + kt0 + vkey) * DD + h * DHH + dh];
      }
      WAIT_ASYNC(0);
    }
#else
#pragma unroll
    for (int it = 0; it < 8; ++it) {
      int i = t + it * 256;
      int key = i >> 6;
      int dh = i & 63;
      size_t g = ((size_t)b * SS + kt0 + key) * DD + h * DHH + dh;
      Kl[key * KSTR + dh] = kb[g];
      Vl[dh * VSTR + key] = vb[g];
    }
#endif
    __syncthreads();

    // S = Q * K^T for two 16-key column tiles
    f32x8 s[2];
#pragma unroll
    for (int nt = 0; nt < 2; ++nt) {
      Frag bk[2];
      const u16* kp = &Kl[(nt * 16 + lr) * KSTR];
      bk[0].q[0] = *(const u32x4*)(kp + halfk);
      bk[0].q[1] = *(const u32x4*)(kp + halfk + 16);
      bk[1].q[0] = *(const u32x4*)(kp + 32 + halfk);
      bk[1].q[1] = *(const u32x4*)(kp + 32 + halfk + 16);
      f32x8 z = zero8();
      z = wmma_bf16(aq[0], bk[0], z);
      z = wmma_bf16(aq[1], bk[1], z);
      s[nt] = z;
    }

    // online softmax in C-fragment layout; write P (bf16) to per-wave LDS
    u16* pl = &Pl[wave * 16 * VSTR];
#pragma unroll
    for (int e = 0; e < 8; ++e) {
      float v0 = s[0][e] * scale;
      float v1 = s[1][e] * scale;
      float rm = fmaxf(v0, v1);
      rm = fmaxf(rm, __shfl_xor(rm, 1, 32));
      rm = fmaxf(rm, __shfl_xor(rm, 2, 32));
      rm = fmaxf(rm, __shfl_xor(rm, 4, 32));
      rm = fmaxf(rm, __shfl_xor(rm, 8, 32));
      float mn = fmaxf(m8[e], rm);
      float corr = __expf(m8[e] - mn);
      float p0 = __expf(v0 - mn);
      float p1 = __expf(v1 - mn);
      float ps = p0 + p1;
      ps += __shfl_xor(ps, 1, 32);
      ps += __shfl_xor(ps, 2, 32);
      ps += __shfl_xor(ps, 4, 32);
      ps += __shfl_xor(ps, 8, 32);
      l8[e] = l8[e] * corr + ps;
      m8[e] = mn;
#pragma unroll
      for (int c = 0; c < 4; ++c) o[c][e] *= corr;
      int row = e + 8 * hi;
      pl[row * VSTR + lr] = f2bf(p0);
      pl[row * VSTR + 16 + lr] = f2bf(p1);
    }

    // P (16x32) as A-fragment, V tiles as B-fragments
    Frag pa;
    {
      const u16* pp = &Pl[wave * 16 * VSTR + lr * VSTR];
      pa.q[0] = *(const u32x4*)(pp + halfk);
      pa.q[1] = *(const u32x4*)(pp + halfk + 16);
    }
#pragma unroll
    for (int c = 0; c < 4; ++c) {
      Frag bv;
      const u16* vp = &Vl[(c * 16 + lr) * VSTR];
      bv.q[0] = *(const u32x4*)(vp + halfk);
      bv.q[1] = *(const u32x4*)(vp + halfk + 16);
      o[c] = wmma_bf16(pa, bv, o[c]);
    }
    __syncthreads();
  }

  // normalize and write ctx in bf16
#pragma unroll
  for (int c = 0; c < 4; ++c)
#pragma unroll
    for (int e = 0; e < 8; ++e) {
      int row = qrow + e + 8 * hi;
      ctx_bf[((size_t)b * SS + row) * DD + h * DHH + c * 16 + lr] = f2bf(o[c][e] / l8[e]);
    }
}

// ---------------- router: scores + top-8 + softmax + recipe mix ----------------
__global__ __launch_bounds__(256) void router_kernel(const float* __restrict__ qbuf,
                                                     const float* __restrict__ nemb,
                                                     const float* __restrict__ smr,
                                                     float* __restrict__ trec) {
  int m = blockIdx.x, t = threadIdx.x;
  __shared__ float qrow[DD];
  __shared__ float red[256];
  __shared__ float sc[NNEUR];
  __shared__ int idx8[TOPK];
  __shared__ float w8[TOPK];

  for (int i = t; i < DD; i += 256) qrow[i] = qbuf[(size_t)m * DD + i];
  __syncthreads();

  int n = t >> 2, part = t & 3;
  float acc = 0.f;
  int d0 = part * 256;
  for (int d = d0; d < d0 + 256; ++d) acc += qrow[d] * nemb[n * DD + d];
  red[t] = acc;
  __syncthreads();
  if (part == 0) sc[n] = red[t] + red[t + 1] + red[t + 2] + red[t + 3];
  __syncthreads();

  if (t == 0) {
    float sl[NNEUR];
    for (int i = 0; i < NNEUR; ++i) sl[i] = sc[i];
    for (int k = 0; k < TOPK; ++k) {
      int bi = 0;
      float bv = -3.4e38f;
      for (int i = 0; i < NNEUR; ++i)
        if (sl[i] > bv) { bv = sl[i]; bi = i; }
      idx8[k] = bi;
      w8[k] = bv;
      sl[bi] = -3.4e38f;
    }
    float mx = w8[0];
    float ssum = 0.f;
    for (int k = 0; k < TOPK; ++k) { w8[k] = __expf(w8[k] - mx); ssum += w8[k]; }
    float inv = 1.0f / ssum;
    for (int k = 0; k < TOPK; ++k) w8[k] *= inv;
  }
  __syncthreads();

  if (t < NBASIS) {
    float a = 0.f;
    for (int k = 0; k < TOPK; ++k) a += w8[k] * smr[idx8[k] * NBASIS + t];
    trec[(size_t)m * NBASIS + t] = a;
  }
}

// ---------------- mix: h = sum_n c_n * y[n,:]; hw = c_n * h_r ----------------
__global__ __launch_bounds__(256) void mix_kernel(const float* __restrict__ ybuf,
                                                  const float* __restrict__ trec,
                                                  u16* __restrict__ hw) {
  int m = blockIdx.x, t = threadIdx.x;
  __shared__ float tr[NBASIS];
  __shared__ float h[RR];
  if (t < NBASIS) tr[t] = trec[(size_t)m * NBASIS + t];
  __syncthreads();
  if (t < RR) {
    float a = 0.f;
#pragma unroll
    for (int n = 0; n < NBASIS; ++n) a += tr[n] * ybuf[(size_t)m * (NBASIS * RR) + n * RR + t];
    h[t] = a;
  }
  __syncthreads();
  for (int i = t; i < NBASIS * RR; i += 256) {
    int n = i >> 6, r = i & 63;
    hw[(size_t)m * (NBASIS * RR) + i] = f2bf(tr[n] * h[r]);
  }
}

// ---------------- host launcher ----------------
extern "C" void kernel_launch(void* const* d_in, const int* in_sizes, int n_in,
                              void* d_out, int out_size, void* d_ws, size_t ws_size,
                              hipStream_t stream) {
  (void)in_sizes; (void)n_in; (void)out_size; (void)ws_size;

  const float* x      = (const float*)d_in[0];
  const float* nrec   = (const float*)d_in[1];
  const float* bemb   = (const float*)d_in[2];
  const float* bA     = (const float*)d_in[3];
  const float* wq     = (const float*)d_in[4];
  const float* bq     = (const float*)d_in[5];
  const float* wk     = (const float*)d_in[6];
  const float* bk     = (const float*)d_in[7];
  const float* wv     = (const float*)d_in[8];
  const float* bv     = (const float*)d_in[9];
  const float* wsm    = (const float*)d_in[10];
  const float* bs     = (const float*)d_in[11];
  const float* w_up   = (const float*)d_in[12];
  const float* b_up   = (const float*)d_in[13];
  const float* w_down = (const float*)d_in[14];
  const float* b_down = (const float*)d_in[15];
  const float* alpha  = (const float*)d_in[16];
  const float* g1     = (const float*)d_in[17];
  const float* be1    = (const float*)d_in[18];
  const float* g2     = (const float*)d_in[19];
  const float* be2    = (const float*)d_in[20];

  char* base = (char*)d_ws;
  size_t off = 0;
  auto take = [&](size_t bytes) -> void* {
    void* p = base + off;
    off = (off + bytes + 255) & ~(size_t)255;
    return p;
  };

  u16* wq_bf   = (u16*)take((size_t)DD * DD * 2);
  u16* wk_bf   = (u16*)take((size_t)DD * DD * 2);
  u16* wv_bf   = (u16*)take((size_t)DD * DD * 2);
  u16* ws_bf   = (u16*)take((size_t)2 * DD * DD * 2);
  u16* wup_bf  = (u16*)take((size_t)DD * DFF * 2);
  u16* wdn_bf  = (u16*)take((size_t)DFF * DD * 2);
  u16* apk     = (u16*)take((size_t)DD * NBASIS * RR * 2);
  u16* apkT    = (u16*)take((size_t)NBASIS * RR * DD * 2);
  float* smr   = (float*)take((size_t)NNEUR * NBASIS * 4);
  float* nemb  = (float*)take((size_t)NNEUR * DD * 4);
  float* xn1_f = (float*)take((size_t)TT * DD * 4);
  u16* xn1_bf  = (u16*)take((size_t)TT * DD * 2);
  float* xn2_f = (float*)take((size_t)TT * DD * 4);
  u16* xn2_bf  = (u16*)take((size_t)TT * DD * 2);
  u16* q_bf    = (u16*)take((size_t)TT * DD * 2);
  u16* k_bf    = (u16*)take((size_t)TT * DD * 2);
  u16* v_bf    = (u16*)take((size_t)TT * DD * 2);
  u16* ctx_bf  = (u16*)take((size_t)TT * DD * 2);
  float* qbuf  = (float*)take((size_t)TT * DD * 4);
  float* trec  = (float*)take((size_t)TT * NBASIS * 4);
  float* ybuf  = (float*)take((size_t)TT * NBASIS * RR * 4);
  u16* hw_bf   = (u16*)take((size_t)TT * NBASIS * RR * 2);
  u16* xfilt   = (u16*)take((size_t)TT * DD * 2);
  u16* hid_bf  = (u16*)take((size_t)TT * DFF * 2);

  // ---- prep ----
  recipe_softmax_kernel<<<1, 64, 0, stream>>>(nrec, smr);
  neuron_emb_kernel<<<NNEUR, 256, 0, stream>>>(smr, bemb, nemb);

  cvt_bf16_kernel<<<1024, 256, 0, stream>>>(wq, wq_bf, DD * DD);
  cvt_bf16_kernel<<<1024, 256, 0, stream>>>(wk, wk_bf, DD * DD);
  cvt_bf16_kernel<<<1024, 256, 0, stream>>>(wv, wv_bf, DD * DD);
  cvt_bf16_kernel<<<1024, 256, 0, stream>>>(wsm, ws_bf, 2 * DD * DD);
  cvt_bf16_kernel<<<1024, 256, 0, stream>>>(w_up, wup_bf, DD * DFF);
  cvt_bf16_kernel<<<1024, 256, 0, stream>>>(w_down, wdn_bf, DFF * DD);
  pack_basis_kernel<<<1024, 256, 0, stream>>>(bA, apk, apkT);

  // ---- layernorms ----
  ln_kernel<<<TT, 256, 0, stream>>>(x, g1, be1, xn1_f, xn1_bf);
  ln_kernel<<<TT, 256, 0, stream>>>(x, g2, be2, xn2_f, xn2_bf);

  // ---- Q/K/V projections (bf16 out) ----
  dim3 blk(256);
  dim3 gQKV(DD / 128, TT / 128);
  gemm_bf16_kernel<EPI_BIAS_BF16><<<gQKV, blk, 0, stream>>>(xn1_bf, wq_bf, bq, nullptr, nullptr, q_bf, TT, DD, DD);
  gemm_bf16_kernel<EPI_BIAS_BF16><<<gQKV, blk, 0, stream>>>(xn1_bf, wk_bf, bk, nullptr, nullptr, k_bf, TT, DD, DD);
  gemm_bf16_kernel<EPI_BIAS_BF16><<<gQKV, blk, 0, stream>>>(xn1_bf, wv_bf, bv, nullptr, nullptr, v_bf, TT, DD, DD);

  // ---- flash attention ----
  attn_kernel<<<dim3(BB * HH, SS / 128), blk, 0, stream>>>(q_bf, k_bf, v_bf, ctx_bf);

  // ---- query = [xn1, ctx] @ ws + bs  (two accumulating GEMMs) ----
  gemm_bf16_kernel<EPI_F32><<<gQKV, blk, 0, stream>>>(xn1_bf, ws_bf, nullptr, nullptr, nullptr, qbuf, TT, DD, DD);
  gemm_bf16_kernel<EPI_BIAS_ADD_F32><<<gQKV, blk, 0, stream>>>(ctx_bf, ws_bf + (size_t)DD * DD, bs, qbuf, nullptr, qbuf, TT, DD, DD);

  // ---- routing ----
  router_kernel<<<TT, 256, 0, stream>>>(qbuf, nemb, smr, trec);

  // ---- y = xn2 @ A_pk  (K=1024, N=2048) ----
  gemm_bf16_kernel<EPI_F32><<<dim3((NBASIS * RR) / 128, TT / 128), blk, 0, stream>>>(
      xn2_bf, apk, nullptr, nullptr, nullptr, ybuf, TT, NBASIS * RR, DD);

  // ---- mix: h, hw ----
  mix_kernel<<<TT, 256, 0, stream>>>(ybuf, trec, hw_bf);

  // ---- delta GEMM + x_filt epilogue: xfilt = bf16(xn2 + alpha*delta) ----
  gemm_bf16_kernel<EPI_XFILT_BF16><<<gQKV, blk, 0, stream>>>(
      hw_bf, apkT, nullptr, xn2_f, alpha, xfilt, TT, DD, NBASIS * RR);

  // ---- FFN up with GELU ----
  gemm_bf16_kernel<EPI_GELU_BF16><<<dim3(DFF / 128, TT / 128), blk, 0, stream>>>(
      xfilt, wup_bf, b_up, nullptr, nullptr, hid_bf, TT, DFF, DD);

  // ---- FFN down + residual -> d_out ----
  gemm_bf16_kernel<EPI_BIAS_ADD_F32><<<gQKV, blk, 0, stream>>>(
      hid_bf, wdn_bf, b_down, x, nullptr, (float*)d_out, TT, DD, DFF);
}
